// GlobalChannelAttention_3015067042107
// MI455X (gfx1250) — matrix operand order repaired and checked by reading, not compiled, not run on hardware
//
#include <hip/hip_runtime.h>
#include <hip/hip_bf16.h>

typedef __attribute__((ext_vector_type(16))) __bf16 v16bf;
typedef __attribute__((ext_vector_type(8)))  float  v8f;

union Frag { uint4 u[2]; v16bf v; };

__device__ __forceinline__ unsigned short f2bf(float f) {
    unsigned int u = __float_as_uint(f);
    u += 0x7fffu + ((u >> 16) & 1u);           // round-to-nearest-even
    return (unsigned short)(u >> 16);
}

// ---------------------------------------------------------------------------
// Kernel 1: global average pool over HW=1024 per (n,c) row. One wave per row.
// ---------------------------------------------------------------------------
__global__ __launch_bounds__(256) void k_gap(const float* __restrict__ x,
                                             float* __restrict__ gap) {
    const int wid  = threadIdx.x >> 5;
    const int lane = threadIdx.x & 31;
    const int row  = blockIdx.x * 8 + wid;                 // 0 .. N*C-1
    const float4* p = reinterpret_cast<const float4*>(x + (size_t)row * 1024);
    float s = 0.f;
#pragma unroll
    for (int i = 0; i < 8; ++i) {
        float4 v = p[i * 32 + lane];
        s += v.x + v.y + v.z + v.w;
    }
#pragma unroll
    for (int m = 16; m >= 1; m >>= 1) s += __shfl_xor(s, m, 32);
    if (lane == 0) gap[row] = s * (1.0f / 1024.0f);
}

// ---------------------------------------------------------------------------
// Kernel 2: per batch: conv1d(K=5,SAME)+sigmoid -> q; softmax(q qT) over all
// C*C entries -> bf16 matrix A (symmetric). One block per batch.
// ---------------------------------------------------------------------------
__global__ __launch_bounds__(256) void k_softmaxA(const float* __restrict__ gap,
                                                  const float* __restrict__ wq,
                                                  const float* __restrict__ bq,
                                                  unsigned short* __restrict__ Ab) {
    __shared__ float qs[512];
    __shared__ float red[256];
    const int n = blockIdx.x, t = threadIdx.x;
    const float* g = gap + n * 512;
    const float w0 = wq[0], w1 = wq[1], w2 = wq[2], w3 = wq[3], w4 = wq[4];
    const float b = bq[0];
    for (int c = t; c < 512; c += 256) {
        float a = b;
        if (c >= 2)   a += w0 * g[c - 2];
        if (c >= 1)   a += w1 * g[c - 1];
        a += w2 * g[c];
        if (c <= 510) a += w3 * g[c + 1];
        if (c <= 509) a += w4 * g[c + 2];
        qs[c] = 1.0f / (1.0f + __expf(-a));
    }
    __syncthreads();
    // max (q > 0, so max of products = (max q)^2)
    red[t] = fmaxf(qs[t], qs[t + 256]);
    __syncthreads();
    for (int s2 = 128; s2 > 0; s2 >>= 1) {
        if (t < s2) red[t] = fmaxf(red[t], red[t + s2]);
        __syncthreads();
    }
    const float M = red[0] * red[0];
    __syncthreads();
    const float q0 = qs[t], q1 = qs[t + 256];
    float partial = 0.f;
    for (int r = 0; r < 512; ++r) {
        const float qr = qs[r];
        partial += __expf(qr * q0 - M) + __expf(qr * q1 - M);
    }
    red[t] = partial;
    __syncthreads();
    for (int s2 = 128; s2 > 0; s2 >>= 1) {
        if (t < s2) red[t] += red[t + s2];
        __syncthreads();
    }
    const float inv = 1.0f / red[0];
    unsigned short* A = Ab + (size_t)n * 262144;
    for (int r = 0; r < 512; ++r) {
        const float qr = qs[r];
        A[r * 512 + t]       = f2bf(__expf(qr * q0 - M) * inv);
        A[r * 512 + t + 256] = f2bf(__expf(qr * q1 - M) * inv);
    }
}

// ---------------------------------------------------------------------------
// Kernel 3: per batch: att^T(512d x 1024p) = A(512x512) @ x(512c x 1024p)
// (valid because A is symmetric), fused with out = x * att^T.
// Workgroup tile 128d x 128p, 8 waves (4x2), wave tile 32d x 64p,
// K-loop over 16 chunks of 32 with v_wmma_f32_16x16x32_bf16.
// ---------------------------------------------------------------------------
#define LSTR 40   // LDS row stride in bf16 elems (80B: 16B-aligned, few conflicts)

__global__ __launch_bounds__(256) void k_attn(const float* __restrict__ x,
                                              const unsigned short* __restrict__ Ab,
                                              float* __restrict__ out) {
    __shared__ unsigned short La[128 * LSTR];  // A rows   [d_local][c_local]
    __shared__ unsigned short Xb[128 * LSTR];  // x tile^T [p_local][c_local]

    const int tid = threadIdx.x;
    const int n = blockIdx.z, d0 = blockIdx.y * 128, p0 = blockIdx.x * 128;
    const int wid = tid >> 5, lane = tid & 31, lr = lane & 15, hi = lane >> 4;
    const int waveM = wid & 3;   // 4 waves along d
    const int waveN = wid >> 2;  // 2 waves along p

    const unsigned short* An = Ab + (size_t)n * 262144;
    const float* xn = x + (size_t)n * 512 * 1024;

    const v8f zz = {0.f, 0.f, 0.f, 0.f, 0.f, 0.f, 0.f, 0.f};
    v8f acc[2][4];
#pragma unroll
    for (int i = 0; i < 2; ++i)
#pragma unroll
        for (int j = 0; j < 4; ++j) acc[i][j] = zz;

    for (int kc = 0; kc < 16; ++kc) {
        const int c0 = kc * 32;
        // Stage 128x32 block of A (bf16, contiguous rows): 512 x 16B chunks.
#pragma unroll
        for (int j = 0; j < 2; ++j) {
            const int id = tid + j * 256;
            const int r = id >> 2, q = id & 3;
            const uint4* src = reinterpret_cast<const uint4*>(
                An + (size_t)(d0 + r) * 512 + c0 + q * 8);
            *reinterpret_cast<uint4*>(&La[r * LSTR + q * 8]) = *src;
        }
        // Stage 32c x 128p tile of x: coalesced float4 loads, transpose+cvt to bf16.
#pragma unroll
        for (int j = 0; j < 4; ++j) {
            const int id = tid + j * 256;
            const int cr = id >> 5, pq = id & 31;
            const float4 v = *reinterpret_cast<const float4*>(
                xn + (size_t)(c0 + cr) * 1024 + p0 + pq * 4);
            Xb[(pq * 4 + 0) * LSTR + cr] = f2bf(v.x);
            Xb[(pq * 4 + 1) * LSTR + cr] = f2bf(v.y);
            Xb[(pq * 4 + 2) * LSTR + cr] = f2bf(v.z);
            Xb[(pq * 4 + 3) * LSTR + cr] = f2bf(v.w);
        }
        __syncthreads();

        // A-operand frags: ISA layout lanes0-15 K{0..7,16..23}, lanes16-31 K{8..15,24..31}
        const int ka = hi ? 8 : 0;
        Frag lf[2];
#pragma unroll
        for (int md = 0; md < 2; ++md) {
            const int row = waveM * 32 + md * 16 + lr;
            lf[md].u[0] = *reinterpret_cast<const uint4*>(&La[row * LSTR + ka]);
            lf[md].u[1] = *reinterpret_cast<const uint4*>(&La[row * LSTR + ka + 16]);
        }
        // B-operand frags: lanes0-15 K{0..7,16..23}, lanes16-31 K{8..15,24..31}
        const int kb = hi ? 8 : 0;
        Frag bfr[4];
#pragma unroll
        for (int np = 0; np < 4; ++np) {
            const int prow = waveN * 64 + np * 16 + lr;
            bfr[np].u[0] = *reinterpret_cast<const uint4*>(&Xb[prow * LSTR + kb]);
            bfr[np].u[1] = *reinterpret_cast<const uint4*>(&Xb[prow * LSTR + kb + 16]);
        }
#pragma unroll
        for (int md = 0; md < 2; ++md)
#pragma unroll
            for (int np = 0; np < 4; ++np)
                acc[md][np] = __builtin_amdgcn_wmma_f32_16x16x32_bf16(
                    false, lf[md].v, false, bfr[np].v,
                    (short)0, acc[md][np], false, false);
        __syncthreads();
    }

    // Epilogue: out[n][d][p] = x[n][d][p] * att^T[d][p]
    // C/D layout: N = lane%16, M = vreg + 8*(lane>=16)
    float* on = out + (size_t)n * 512 * 1024;
#pragma unroll
    for (int md = 0; md < 2; ++md)
#pragma unroll
        for (int np = 0; np < 4; ++np)
#pragma unroll
            for (int v = 0; v < 8; ++v) {
                const int gd = d0 + waveM * 32 + md * 16 + v + hi * 8;
                const int gp = p0 + waveN * 64 + np * 16 + lr;
                const size_t idx = (size_t)gd * 1024 + gp;
                on[idx] = xn[idx] * acc[md][np][v];
            }
}

// ---------------------------------------------------------------------------
extern "C" void kernel_launch(void* const* d_in, const int* in_sizes, int n_in,
                              void* d_out, int out_size, void* d_ws, size_t ws_size,
                              hipStream_t stream) {
    const float* x  = (const float*)d_in[0];   // (32,512,32,32) fp32
    const float* wq = (const float*)d_in[1];   // (1,1,5)
    const float* bq = (const float*)d_in[2];   // (1,)
    float* out = (float*)d_out;                // (32,512,32,32) fp32

    float* gap = (float*)d_ws;                                         // 64 KB
    unsigned short* Ab = (unsigned short*)((char*)d_ws + 65536);       // 16 MB bf16

    k_gap<<<2048, 256, 0, stream>>>(x, gap);                 // 16384 rows / 8 per block
    k_softmaxA<<<32, 256, 0, stream>>>(gap, wq, bq, Ab);     // one block per batch
    dim3 g3(8, 4, 32);                                       // (p-blocks, d-blocks, N)
    k_attn<<<g3, 256, 0, stream>>>(x, Ab, out);
}
